// Encoder_33406255628854
// MI455X (gfx1250) — matrix operand order repaired and checked by reading, not compiled, not run on hardware
//
#include <hip/hip_runtime.h>
#include <hip/hip_fp16.h>

// ---------------- problem constants ----------------
constexpr int B_  = 64;
constexpr int S_  = 400;
constexpr int D_  = 128;
constexpr int L_  = 4;
constexpr int KW_ = 7;
constexpr int H_  = 8;
constexpr int M_  = B_ * S_;      // 25600 token rows
constexpr int MT_ = M_ / 16;      // 1600 row tiles
constexpr int ST_ = S_ / 16;      // 25 query tiles per batch row

typedef __attribute__((ext_vector_type(16))) _Float16 v16h;
typedef __attribute__((ext_vector_type(8)))  float    v8f;

union Frag { v16h h; unsigned u[8]; };

__device__ __forceinline__ unsigned pack2f(float a, float b) {
  return (unsigned)__half_as_ushort(__float2half(a)) |
         ((unsigned)__half_as_ushort(__float2half(b)) << 16);
}
__device__ __forceinline__ v8f wmma16(const Frag& a, const Frag& b, v8f c) {
  return __builtin_amdgcn_wmma_f32_16x16x32_f16(false, a.h, false, b.h,
                                                (short)0, c, false, false);
}
__device__ __forceinline__ void frag_from2(Frag& f, uint4 a, uint4 b) {
  f.u[0] = a.x; f.u[1] = a.y; f.u[2] = a.z; f.u[3] = a.w;
  f.u[4] = b.x; f.u[5] = b.y; f.u[6] = b.z; f.u[7] = b.w;
}

// ---------------- kernel: x + sinusoidal positional encoding -> fp32 ----------------
__global__ void enc_posenc(const float* __restrict__ x, float* __restrict__ out) {
  size_t i = (size_t)blockIdx.x * 256 + threadIdx.x;   // M_*D_ threads
  int d = (int)(i & (D_ - 1));
  size_t bs = i >> 7;
  int s = (int)(bs % S_);
  int c = d & 63;
  float freq = __expf(-(2.0f * (float)c / 128.0f) * 9.2103403719761836f); // ln(1e4)
  float ang  = (float)s * freq;
  float pe   = (d < 64) ? sinf(ang) : cosf(ang);
  out[i] = x[i] + pe;
}

// ---------------- kernel: row LayerNorm (fp32 in) -> f16 out; 1 wave per row ----------------
__global__ void enc_ln16(const float* __restrict__ in, const float* __restrict__ g,
                         const float* __restrict__ bt, __half* __restrict__ out) {
  int row  = blockIdx.x * 8 + (threadIdx.x >> 5);
  int lane = threadIdx.x & 31;
  const float4 xv = ((const float4*)(in + (size_t)row * D_))[lane];
  float s  = xv.x + xv.y + xv.z + xv.w;
  float s2 = xv.x*xv.x + xv.y*xv.y + xv.z*xv.z + xv.w*xv.w;
  s  += __shfl_xor(s, 1);  s  += __shfl_xor(s, 2);  s  += __shfl_xor(s, 4);
  s  += __shfl_xor(s, 8);  s  += __shfl_xor(s, 16);
  s2 += __shfl_xor(s2, 1); s2 += __shfl_xor(s2, 2); s2 += __shfl_xor(s2, 4);
  s2 += __shfl_xor(s2, 8); s2 += __shfl_xor(s2, 16);
  float mean = s * (1.0f / 128.0f);
  float var  = s2 * (1.0f / 128.0f) - mean * mean;
  float rstd = rsqrtf(var + 1e-5f);
  const float4 gv = ((const float4*)g)[lane];
  const float4 bv = ((const float4*)bt)[lane];
  float y0 = (xv.x - mean) * rstd * gv.x + bv.x;
  float y1 = (xv.y - mean) * rstd * gv.y + bv.y;
  float y2 = (xv.z - mean) * rstd * gv.z + bv.z;
  float y3 = (xv.w - mean) * rstd * gv.w + bv.w;
  uint2 st; st.x = pack2f(y0, y1); st.y = pack2f(y2, y3);
  ((uint2*)(out + (size_t)row * D_))[lane] = st;
}

// ------- kernel: repack [128f x 128d] fp32 weight -> f16 pairs, row-contiguous [f][d2] -------
__global__ void enc_repack_mat(const float* __restrict__ w, unsigned* __restrict__ dst,
                               float scale) {
  int i  = blockIdx.x * 256 + threadIdx.x;   // 8192
  int f  = i >> 6, d2 = i & 63;
  dst[i] = pack2f(w[f * 128 + 2 * d2] * scale, w[f * 128 + 2 * d2 + 1] * scale);
}

// ------- kernel: repack conv weights [f][d][k] -> f16 pairs [f][kw*64 + d2] (448/row) -------
__global__ void enc_repack_conv(const float* __restrict__ w, unsigned* __restrict__ dst) {
  int i  = blockIdx.x * 256 + threadIdx.x;   // 128*448 = 57344
  int f  = i / 448;
  int p  = i % 448;
  int kw = p >> 6, d2 = p & 63;
  dst[i] = pack2f(w[(f * 128 + 2 * d2) * KW_ + kw],
                  w[(f * 128 + 2 * d2 + 1) * KW_ + kw]);
}

// ---------------- kernel: conv as implicit GEMM  (M tile 16 x N 128, K = 7*128) ----------------
__global__ void enc_conv_gemm(const __half* __restrict__ h16, const unsigned* __restrict__ Wc,
                              const float* __restrict__ bias, __half* __restrict__ out16) {
  __shared__ __align__(16) unsigned sA[22 * 64];   // 22 seq rows (16 + halo) x 128 f16
  const int tid = threadIdx.x;
  const int blk = blockIdx.x;
  const int b   = blk / ST_;
  const int s0  = (blk % ST_) * 16;
  const unsigned* gh = (const unsigned*)h16;
  for (int i = tid; i < 22 * 64; i += 256) {
    int row = i >> 6, cp = i & 63;
    int sr  = s0 + row - 3;
    unsigned v = 0u;
    if (sr >= 0 && sr < S_) v = gh[((size_t)b * S_ + sr) * 64 + cp];
    sA[i] = v;
  }
  __syncthreads();
  const int lane = tid & 31, wave = tid >> 5;
  const int n0 = wave * 16, mrow = lane & 15, hh = lane >> 4;
  const unsigned* brow = Wc + (size_t)(n0 + mrow) * 448;   // this lane's output channel
  v8f c;
#pragma unroll
  for (int i = 0; i < 8; ++i) c[i] = 0.0f;
#pragma unroll 4
  for (int chunk = 0; chunk < 28; ++chunk) {
    const int kb = chunk * 32;
    const int kw = kb >> 7, dbase = kb & 127;
    if (chunk + 2 < 28) {
      const int kbn = kb + 64;
      __builtin_prefetch(brow + ((kbn >> 7) * 64 + ((kbn & 127) >> 1) + hh * 8), 0, 1);
    }
    Frag a, bf;
#pragma unroll
    for (int v = 0; v < 8; ++v) {
      const int kA = 2 * v + hh * 16;   // contiguous k = 16h + s
      a.u[v] = sA[(mrow + kw) * 64 + ((dbase + kA) >> 1)];
    }
    const uint4* bp4 = (const uint4*)(brow + (kw * 64 + (dbase >> 1) + hh * 8));
    frag_from2(bf, bp4[0], bp4[1]);                                 // 2x b128 per fragment
    c = wmma16(a, bf, c);
  }
#pragma unroll
  for (int r = 0; r < 8; ++r) {
    const int row = s0 + r + hh * 8;
    const int col = n0 + mrow;
    out16[((size_t)b * S_ + row) * D_ + col] = __float2half(c[r] + bias[col]);
  }
}

// ------- kernel: generic M x 128 x 128 GEMM; async LDS staging; optional V-transpose out -------
__global__ void enc_gemm128(const __half* __restrict__ A, const unsigned* __restrict__ Bw,
                            const float* __restrict__ bias, float* __restrict__ out32,
                            __half* __restrict__ out16, int relu, float bscale, int vtrans) {
  __shared__ __align__(16) unsigned sA[16 * 64];
  const int tid = threadIdx.x;
  const int m0  = blockIdx.x * 16;
  {
    // CDNA5 async copy: global -> LDS, no VGPR round trip (ASYNCcnt tracked)
    unsigned ldsAddr = (unsigned)(size_t)(void*)sA + (unsigned)tid * 16u;
    unsigned long long ga =
        (unsigned long long)(const void*)(A + (size_t)m0 * D_) + (size_t)tid * 16u;
    asm volatile("global_load_async_to_lds_b128 %0, %1, off"
                 :: "v"(ldsAddr), "v"(ga) : "memory");
    asm volatile("s_wait_asynccnt 0x0" ::: "memory");
  }
  __syncthreads();
  const int lane = tid & 31, wave = tid >> 5;
  const int n0 = wave * 16, mrow = lane & 15, hh = lane >> 4;
  const unsigned* brow = Bw + (size_t)(n0 + mrow) * 64;
  v8f c;
#pragma unroll
  for (int i = 0; i < 8; ++i) c[i] = 0.0f;
#pragma unroll
  for (int chunk = 0; chunk < 4; ++chunk) {
    const int kb = chunk * 32;
    Frag a, bf;
#pragma unroll
    for (int v = 0; v < 8; ++v) {
      const int kA = 2 * v + hh * 16;
      a.u[v] = sA[mrow * 64 + ((kb + kA) >> 1)];
    }
    const uint4* bp4 = (const uint4*)(brow + ((kb >> 1) + hh * 8));
    frag_from2(bf, bp4[0], bp4[1]);
    c = wmma16(a, bf, c);
  }
#pragma unroll
  for (int r = 0; r < 8; ++r) {
    const int row = m0 + r + hh * 8;
    const int col = n0 + mrow;
    float val = c[r] + bias[col] * bscale;
    if (relu) val = fmaxf(val, 0.0f);
    if (out32) out32[(size_t)row * D_ + col] = val;
    if (out16) {
      if (vtrans) {   // V projection: write [b][h][hd][s] for flash PV b128 loads
        const int bb = row / S_, ss = row % S_;
        const int hq = col >> 4, hd = col & 15;
        out16[(((size_t)bb * H_ + hq) * 16 + hd) * S_ + ss] = __float2half(val);
      } else {
        out16[(size_t)row * D_ + col] = __float2half(val);
      }
    }
  }
}

// ---- kernel: fused causal flash attention; 32 k-positions/iter, full-K PV WMMA ----
// q16 pre-scaled by 1/sqrt(hd). vT layout [b][h][hd][s].
__global__ void enc_flash_attn(const __half* __restrict__ q16, const __half* __restrict__ k16,
                               const __half* __restrict__ vT, __half* __restrict__ o16) {
  __shared__ __align__(16) __half sP[16 * 32];
  const int qt = blockIdx.x, hgrp = blockIdx.y, b = blockIdx.z;
  const int hd0 = hgrp * 16;
  const int q0  = qt * 16;
  const int lane = threadIdx.x, mrow = lane & 15, hh = lane >> 4;
  const unsigned* qg = (const unsigned*)q16;
  const unsigned* kg = (const unsigned*)k16;
  const unsigned* vg = (const unsigned*)vT;

  Frag qa;                                   // A-frag: 16 q rows x hd(16, zero-padded to 32)
  {
    if (hh == 0) {
      const uint4* qp = (const uint4*)(qg + ((size_t)b * S_ + q0 + mrow) * 64 + (hd0 >> 1));
      frag_from2(qa, qp[0], qp[1]);
    } else {
      for (int v = 0; v < 8; ++v) qa.u[v] = 0u;
    }
  }
  float mr[8], lr[8];
  v8f o;
#pragma unroll
  for (int i = 0; i < 8; ++i) { mr[i] = -1e30f; lr[i] = 0.0f; o[i] = 0.0f; }

  const int T = qt + 1;                      // causal 16-tile count
  const int P = (T + 1) >> 1;                // 32-wide iterations
  for (int p = 0; p < P; ++p) {
    const int k0 = p * 32;
    const bool v1 = (2 * p + 1) < T;         // second 16-tile valid (uniform)

    Frag kb0, kb1;
    if (hh == 0) {                           // score B-frag: K^T, hd rows (0..15)
      const uint4* kp = (const uint4*)(kg + ((size_t)b * S_ + k0 + mrow) * 64 + (hd0 >> 1));
      frag_from2(kb0, kp[0], kp[1]);
    } else {
#pragma unroll
      for (int v = 0; v < 8; ++v) kb0.u[v] = 0u;
    }
    v8f s0, s1;
#pragma unroll
    for (int i = 0; i < 8; ++i) { s0[i] = 0.0f; s1[i] = 0.0f; }
    s0 = wmma16(qa, kb0, s0);
    if (v1) {
      if (hh == 0) {
        const uint4* kp = (const uint4*)(kg + ((size_t)b * S_ + k0 + 16 + mrow) * 64 + (hd0 >> 1));
        frag_from2(kb1, kp[0], kp[1]);
      } else {
#pragma unroll
        for (int v = 0; v < 8; ++v) kb1.u[v] = 0u;
      }
      s1 = wmma16(qa, kb1, s1);
    }

#pragma unroll
    for (int r = 0; r < 8; ++r) {            // online softmax over 32 columns
      const int qpos = q0 + r + hh * 8;
      float sv0 = s0[r];
      if (2 * p == qt && (k0 + mrow > qpos)) sv0 = -1e30f;
      float sv1 = v1 ? s1[r] : -1e30f;
      if (v1 && (2 * p + 1 == qt) && (k0 + 16 + mrow > qpos)) sv1 = -1e30f;
      float rmax = fmaxf(sv0, sv1);
      rmax = fmaxf(rmax, __shfl_xor(rmax, 1));
      rmax = fmaxf(rmax, __shfl_xor(rmax, 2));
      rmax = fmaxf(rmax, __shfl_xor(rmax, 4));
      rmax = fmaxf(rmax, __shfl_xor(rmax, 8));
      const float mnew = fmaxf(mr[r], rmax);
      const float p0 = __expf(sv0 - mnew);
      const float p1 = __expf(sv1 - mnew);
      float rsum = p0 + p1;
      rsum += __shfl_xor(rsum, 1);
      rsum += __shfl_xor(rsum, 2);
      rsum += __shfl_xor(rsum, 4);
      rsum += __shfl_xor(rsum, 8);
      const float corr = __expf(mr[r] - mnew);
      lr[r] = lr[r] * corr + rsum;
      o[r] *= corr;
      mr[r] = mnew;
      sP[(r + hh * 8) * 32 + mrow]      = __float2half(p0);   // C-layout -> LDS bounce
      sP[(r + hh * 8) * 32 + 16 + mrow] = __float2half(p1);
    }
    __syncthreads();

    Frag pa;                                 // P in A-frag layout, full K=32
    const unsigned* sPu = (const unsigned*)sP;
#pragma unroll
    for (int v = 0; v < 8; ++v) {
      const int kL2 = v + 8 * hh;   // kLocal/2
      pa.u[v] = sPu[mrow * 16 + kL2];
    }

    Frag vb;                                 // V B-frag: [hd][s] rows -> contiguous pairs
    if (hh == 0 || v1) {
      const uint4* vp = (const uint4*)(vg + ((size_t)(b * H_ + hgrp) * 16 + mrow) * (S_ / 2)
                                          + (k0 >> 1) + hh * 8);
      frag_from2(vb, vp[0], vp[1]);
    } else {
#pragma unroll
      for (int v = 0; v < 8; ++v) vb.u[v] = 0u;
    }
    o = wmma16(pa, vb, o);
    __syncthreads();
  }
#pragma unroll
  for (int r = 0; r < 8; ++r) {
    const int row = q0 + r + hh * 8;
    o16[((size_t)b * S_ + row) * D_ + hd0 + mrow] = __float2half(o[r] / lr[r]);
  }
}

// ---------------- host orchestration ----------------
extern "C" void kernel_launch(void* const* d_in, const int* in_sizes, int n_in,
                              void* d_out, int out_size, void* d_ws, size_t ws_size,
                              hipStream_t stream) {
  (void)in_sizes; (void)n_in; (void)out_size; (void)ws_size;
  const float* x      = (const float*)d_in[0];
  const float* ln_g   = (const float*)d_in[1];
  const float* ln_b   = (const float*)d_in[2];
  const float* dw_w   = (const float*)d_in[3];
  const float* dw_b   = (const float*)d_in[4];
  const float* pw_w   = (const float*)d_in[5];
  const float* pw_b   = (const float*)d_in[6];
  const float* attln_g= (const float*)d_in[7];
  const float* attln_b= (const float*)d_in[8];
  const float* wq     = (const float*)d_in[9];
  const float* bq     = (const float*)d_in[10];
  const float* wk     = (const float*)d_in[11];
  const float* bk     = (const float*)d_in[12];
  const float* wv     = (const float*)d_in[13];
  const float* bv     = (const float*)d_in[14];
  const float* wp     = (const float*)d_in[15];
  const float* bp     = (const float*)d_in[16];
  const float* ffnln_g= (const float*)d_in[17];
  const float* ffnln_b= (const float*)d_in[18];
  const float* fc_w   = (const float*)d_in[19];
  const float* fc_b   = (const float*)d_in[20];

  // workspace carve-out (all regions 256B aligned)
  char* ws = (char*)d_ws;
  size_t off = 0;
  auto carve = [&](size_t bytes) {
    void* p = ws + off;
    off = (off + bytes + 255) & ~(size_t)255;
    return p;
  };
  float*    bufOut = (float*)   carve((size_t)M_ * D_ * 4);   // running fp32 stream
  __half*   h16    = (__half*)  carve((size_t)M_ * D_ * 2);   // LN outputs f16
  __half*   c16    = (__half*)  carve((size_t)M_ * D_ * 2);   // conv out / attn out f16
  __half*   q16    = (__half*)  carve((size_t)M_ * D_ * 2);
  __half*   k16    = (__half*)  carve((size_t)M_ * D_ * 2);
  __half*   v16    = (__half*)  carve((size_t)M_ * D_ * 2);   // transposed [b][h][hd][s]
  unsigned* wConv  = (unsigned*)carve((size_t)L_ * KW_ * 64 * 128 * 4);
  unsigned* wMat   = (unsigned*)carve((size_t)9 * 64 * 128 * 4);
  const int PW = 64 * 128;   // 8192 u32 per packed 128x128 matrix
  const int CW = KW_ * 64 * 128;

  // weight repacks (every call: deterministic)
  for (int l = 0; l < L_; ++l) {
    enc_repack_conv<<<CW / 256, 256, 0, stream>>>(dw_w + (size_t)l * D_ * D_ * KW_, wConv + (size_t)l * CW);
    enc_repack_mat <<<PW / 256, 256, 0, stream>>>(pw_w + (size_t)l * D_ * D_, wMat + (size_t)l * PW, 1.0f);
  }
  enc_repack_mat<<<PW / 256, 256, 0, stream>>>(wq,   wMat + 4 * PW, 0.25f);  // fold 1/sqrt(hd)
  enc_repack_mat<<<PW / 256, 256, 0, stream>>>(wk,   wMat + 5 * PW, 1.0f);
  enc_repack_mat<<<PW / 256, 256, 0, stream>>>(wv,   wMat + 6 * PW, 1.0f);
  enc_repack_mat<<<PW / 256, 256, 0, stream>>>(wp,   wMat + 7 * PW, 1.0f);
  enc_repack_mat<<<PW / 256, 256, 0, stream>>>(fc_w, wMat + 8 * PW, 1.0f);

  // x + positional encoding
  enc_posenc<<<(M_ * D_) / 256, 256, 0, stream>>>(x, bufOut);

  // conv stack: LN -> implicit-GEMM conv -> pointwise GEMM
  for (int l = 0; l < L_; ++l) {
    enc_ln16<<<M_ / 8, 256, 0, stream>>>(bufOut, ln_g + l * D_, ln_b + l * D_, h16);
    enc_conv_gemm<<<MT_, 256, 0, stream>>>(h16, wConv + (size_t)l * CW, dw_b + l * D_, c16);
    enc_gemm128<<<MT_, 256, 0, stream>>>(c16, wMat + (size_t)l * PW, pw_b + l * D_,
                                         bufOut, nullptr, 0, 1.0f, 0);
  }

  // attention: LN -> Q/K/V -> fused flash -> output projection
  enc_ln16<<<M_ / 8, 256, 0, stream>>>(bufOut, attln_g, attln_b, h16);
  enc_gemm128<<<MT_, 256, 0, stream>>>(h16, wMat + 4 * PW, bq, nullptr, q16, 0, 0.25f, 0);
  enc_gemm128<<<MT_, 256, 0, stream>>>(h16, wMat + 5 * PW, bk, nullptr, k16, 0, 1.0f, 0);
  enc_gemm128<<<MT_, 256, 0, stream>>>(h16, wMat + 6 * PW, bv, nullptr, v16, 0, 1.0f, 1);
  enc_flash_attn<<<dim3(ST_, H_, B_), 32, 0, stream>>>(q16, k16, v16, c16);
  enc_gemm128<<<MT_, 256, 0, stream>>>(c16, wMat + 7 * PW, bp, bufOut, nullptr, 0, 1.0f, 0);

  // ffn: LN -> FC + ReLU -> final fp32 output
  enc_ln16<<<M_ / 8, 256, 0, stream>>>(bufOut, ffnln_g, ffnln_b, h16);
  enc_gemm128<<<MT_, 256, 0, stream>>>(h16, wMat + 8 * PW, fc_b, (float*)d_out, nullptr, 1, 1.0f, 0);
}
